// VectorQuantizer_32916629356739
// MI455X (gfx1250) — matrix operand order repaired and checked by reading, not compiled, not run on hardware
//
#include <hip/hip_runtime.h>
#include <hip/hip_bf16.h>
#include <stddef.h>

// Problem constants
#define BB 32
#define CC 64        // = D (embedding dim)
#define HH 64
#define WW 64
#define KK 1024
#define NN (BB * HH * WW)        // 131072 rows
#define HW (HH * WW)             // 4096
#define NC ((size_t)NN * CC)     // 8388608 elements in x / quantized

// d_out float layout (concatenated reference outputs):
#define OUT_LOSS 0
#define OUT_KL   1
#define OUT_Q    33
#define OUT_E    (33 + 8388608)  // 8388641 (odd -> only 4B-aligned)

// d_ws byte layout:
#define WS_WSQ     0                       // 1024 f32   (4 KB)
#define WS_PARTIAL 4096                    // 512  f32   (2 KB)
#define WS_PACKW   8192                    // 64*2*32*16 bf16 = 128 KB
#define WS_IDX     (8192 + 131072)         // 131072 i32 = 512 KB
#define WS_NEEDED  (WS_IDX + 4 * NN)

typedef __attribute__((ext_vector_type(16))) __bf16 v16bf;
typedef __attribute__((ext_vector_type(8)))  float  v8f;

__device__ __forceinline__ unsigned short f2bf(float f) {
    union { float f; unsigned u; } x; x.f = f;
    unsigned r = x.u + 0x7FFFu + ((x.u >> 16) & 1u);   // round-to-nearest-even
    return (unsigned short)(r >> 16);
}

// ---------------------------------------------------------------------------
// ||w_k||^2 for all K codes
__global__ void vq_wsq_kernel(const float* __restrict__ w, float* __restrict__ wsq) {
    int k = blockIdx.x * blockDim.x + threadIdx.x;   // 0..1023
    const float* row = w + k * CC;
    float s = 0.f;
#pragma unroll 8
    for (int d = 0; d < CC; ++d) { float v = row[d]; s += v * v; }
    wsq[k] = s;
}

// ---------------------------------------------------------------------------
// Pre-pack codebook into bf16 WMMA B-matrix lane layout:
// B is 32(K=d-slice) x 16(N=codes).  Per ISA: lane L holds column n = L&15,
// K rows kbase..kbase+7 and 16+kbase..16+kbase+7 (kbase = 0 for lanes 0-15, 8 for lanes 16-31),
// 2 halves per VGPR -> 16 contiguous halves per lane.
__global__ void vq_packw_kernel(const float* __restrict__ w, unsigned short* __restrict__ pk) {
    int t = blockIdx.x * blockDim.x + threadIdx.x;   // 0..4095 = tile*2*32 + slice*32 + lane
    int lane  = t & 31;
    int slice = (t >> 5) & 1;                        // which 32-wide d slice
    int tile  = t >> 6;                              // code tile 0..63
    int n     = lane & 15;
    int kbase = (lane < 16) ? 0 : 8;
    int code  = tile * 16 + n;
    unsigned short* dst = pk + (size_t)t * 16;
#pragma unroll
    for (int i = 0; i < 16; ++i)
        dst[i] = f2bf(w[code * CC + slice * 32 + kbase + i + (i & 8)]);
}

// ---------------------------------------------------------------------------
// encodings = one_hot(x_tilde)  [N, K]  (536 MB stream; scalar B32 stores
// because the region base is only 4B aligned)
__global__ void vq_enc_kernel(const int* __restrict__ xt, float* __restrict__ oute) {
    int wid  = (blockIdx.x * blockDim.x + threadIdx.x) >> 5;  // row n
    int lane = threadIdx.x & 31;
    int idx  = xt[wid];
    float* row = oute + (size_t)wid * KK;
#pragma unroll
    for (int t = 0; t < 32; ++t) {
        int col = t * 32 + lane;
        row[col] = (col == idx) ? 1.0f : 0.0f;
    }
}

// ---------------------------------------------------------------------------
// quantized[b,c,h,w] = weight[x_tilde[n], c];  partial sums of (q - x)^2
__global__ void vq_quant_loss_kernel(const float* __restrict__ in1,
                                     const float* __restrict__ w,
                                     const int*   __restrict__ xt,
                                     float* __restrict__ outq,
                                     float* __restrict__ partial) {
    int n   = blockIdx.x * blockDim.x + threadIdx.x;  // 0..N-1
    int b   = n >> 12;                                 // /4096
    int hw  = n & 4095;
    int idx = xt[n];
    const float* wrow = w + idx * CC;
    const float* xp   = in1  + (size_t)b * CC * HW + hw;
    float*       qp   = outq + (size_t)b * CC * HW + hw;
    float acc = 0.f;
#pragma unroll 8
    for (int c = 0; c < CC; ++c) {
        float q = wrow[c];
        float x = xp[(size_t)c * HW];
        float d = q - x;
        acc += d * d;
        qp[(size_t)c * HW] = q;
    }
    __shared__ float red[256];
    int tid = threadIdx.x;
    red[tid] = acc;
    __syncthreads();
    for (int s = 128; s > 0; s >>= 1) {
        if (tid < s) red[tid] += red[tid + s];
        __syncthreads();
    }
    if (tid == 0) partial[blockIdx.x] = red[0];
}

// ---------------------------------------------------------------------------
// loss = 1.25 * sum / (N*C);  copy KL passthrough.  Fixed-order -> deterministic.
__global__ void vq_finalize_kernel(const float* __restrict__ partial,
                                   const float* __restrict__ kl,
                                   float* __restrict__ out) {
    __shared__ float red[256];
    int t = threadIdx.x;
    red[t] = partial[t] + partial[t + 256];
    __syncthreads();
    for (int s = 128; s > 0; s >>= 1) {
        if (t < s) red[t] += red[t + s];
        __syncthreads();
    }
    if (t == 0) out[OUT_LOSS] = 1.25f * red[0] / (float)NC;
    if (t < 32) out[OUT_KL + t] = kl[t];
}

// ---------------------------------------------------------------------------
// Distance matrix + argmin via bf16 WMMA (reference-faithful path; result
// cancels out of the outputs, so it lands in scratch).  One wave per 16-row
// tile; loops 64 code tiles, 2x v_wmma_f32_16x16x32_bf16 per tile (K=D=64).
__global__ void __launch_bounds__(128)
vq_argmin_wmma_kernel(const float* __restrict__ in1,
                      const unsigned short* __restrict__ packW,
                      const float* __restrict__ wsq,
                      int* __restrict__ outidx) {
    __shared__ float xt_s[4][16][65];   // [wave][row][d], pad to kill conflicts
    __shared__ float xsq_s[4][16];
    __shared__ float rv[4][16][17];
    __shared__ int   ri[4][16][17];

    int lane = threadIdx.x & 31;
    int wv   = threadIdx.x >> 5;                 // 0..3
    int tileRow = blockIdx.x * 4 + wv;           // 0..8191
    int n0 = tileRow * 16;
    int b  = n0 >> 12;
    int hw0 = n0 & 4095;                         // 16 rows never cross b or h

    float* xt = &xt_s[wv][0][0];
    const float* xbase = in1 + (size_t)b * CC * HW + hw0;

    // Stage x tile (16 rows x 64 dims) through LDS: flat[n][d] = input1[b,d,h,w]
#pragma unroll 4
    for (int t = 0; t < 32; ++t) {
        int q = t * 32 + lane;                   // q = d*16 + r
        int d = q >> 4, r = q & 15;
        xt[r * 65 + d] = xbase[(size_t)d * HW + r];
    }
    __syncthreads();

    if (lane < 16) {
        float s = 0.f;
#pragma unroll 8
        for (int d = 0; d < CC; ++d) { float v = xt[lane * 65 + d]; s += v * v; }
        xsq_s[wv][lane] = s;
    }
    __syncthreads();

    // Build A operands per ISA 16-bit A 16x32 layout:
    // lanes 0-15: M=lane, K halves {0..7, 16..23}; lanes 16-31: M=lane-16, {8..15, 24..31}
    int m = lane & 15;
    int koff = (lane < 16) ? 0 : 8;
    union Ubf { unsigned short u[16]; v16bf v; } ua0, ua1;
#pragma unroll
    for (int i = 0; i < 8; ++i) {
        ua0.u[i]     = f2bf(xt[m * 65 + koff + i]);
        ua0.u[8 + i] = f2bf(xt[m * 65 + 16 + koff + i]);
        ua1.u[i]     = f2bf(xt[m * 65 + 32 + koff + i]);
        ua1.u[8 + i] = f2bf(xt[m * 65 + 48 + koff + i]);
    }

    float bestv[8]; int besti[8]; float xs[8];
    int mbase = (lane < 16) ? 0 : 8;
#pragma unroll
    for (int j = 0; j < 8; ++j) {
        bestv[j] = 3.4e38f; besti[j] = 0;
        xs[j] = xsq_s[wv][mbase + j];
    }

    for (int nt = 0; nt < 64; ++nt) {
        union Ubf ub0, ub1;
        const float4* bp0 = (const float4*)(packW + ((size_t)(nt * 2 + 0) * 32 + lane) * 16);
        const float4* bp1 = (const float4*)(packW + ((size_t)(nt * 2 + 1) * 32 + lane) * 16);
        ((float4*)ub0.u)[0] = bp0[0]; ((float4*)ub0.u)[1] = bp0[1];
        ((float4*)ub1.u)[0] = bp1[0]; ((float4*)ub1.u)[1] = bp1[1];

        v8f c = {};
#if defined(__HIP_DEVICE_COMPILE__)
        c = __builtin_amdgcn_wmma_f32_16x16x32_bf16(false, ua0.v, false, ub0.v,
                                                    (short)0, c, false, false);
        c = __builtin_amdgcn_wmma_f32_16x16x32_bf16(false, ua1.v, false, ub1.v,
                                                    (short)0, c, false, false);
#endif
        // C layout: lane holds column n = lane&15; c[j] is row mbase+j
        float wn  = wsq[nt * 16 + m];
        int  nidx = nt * 16 + m;
#pragma unroll
        for (int j = 0; j < 8; ++j) {
            float dist = xs[j] + wn - 2.0f * c[j];
            if (dist < bestv[j]) { bestv[j] = dist; besti[j] = nidx; }
        }
    }

    // Cross-lane argmin per row (first-index tie-break, matching jnp.argmin)
#pragma unroll
    for (int j = 0; j < 8; ++j) {
        rv[wv][mbase + j][m] = bestv[j];
        ri[wv][mbase + j][m] = besti[j];
    }
    __syncthreads();
    if (lane < 16) {
        float bv = rv[wv][lane][0]; int bi = ri[wv][lane][0];
#pragma unroll
        for (int c2 = 1; c2 < 16; ++c2) {
            float v = rv[wv][lane][c2]; int i2 = ri[wv][lane][c2];
            if (v < bv || (v == bv && i2 < bi)) { bv = v; bi = i2; }
        }
        outidx[n0 + lane] = bi;
    }
}

// ---------------------------------------------------------------------------
extern "C" void kernel_launch(void* const* d_in, const int* in_sizes, int n_in,
                              void* d_out, int out_size, void* d_ws, size_t ws_size,
                              hipStream_t stream) {
    const float* in1  = (const float*)d_in[0];   // input1   [B,C,H,W] f32
    const float* kl   = (const float*)d_in[1];   // input2_KL [32] f32
    const float* w    = (const float*)d_in[2];   // weight   [K,C] f32
    const int*   xt   = (const int*)  d_in[3];   // x_tilde  [N] i32
    float* out = (float*)d_out;

    char* ws = (char*)d_ws;
    float*          ws_wsq     = (float*)(ws + WS_WSQ);
    float*          ws_partial = (float*)(ws + WS_PARTIAL);
    unsigned short* ws_packw   = (unsigned short*)(ws + WS_PACKW);
    int*            ws_idx     = (int*)(ws + WS_IDX);

    const bool do_argmin = ws_size >= (size_t)WS_NEEDED;  // constant per run -> deterministic

    // Live outputs (memory-bound path, exact fp32):
    vq_enc_kernel<<<NN / 8, 256, 0, stream>>>(xt, out + OUT_E);
    vq_quant_loss_kernel<<<NN / 256, 256, 0, stream>>>(in1, w, xt, out + OUT_Q, ws_partial);
    vq_finalize_kernel<<<1, 256, 0, stream>>>(ws_partial, kl, out);

    // Reference-faithful distance GEMM + argmin via bf16 WMMA (hides under the
    // ~26us one-hot write at PFLOPS-class matrix rate):
    if (do_argmin) {
        vq_wsq_kernel<<<KK / 256, 256, 0, stream>>>(w, ws_wsq);
        vq_packw_kernel<<<16, 256, 0, stream>>>(w, ws_packw);
        vq_argmin_wmma_kernel<<<NN / 16 / 4, 128, 0, stream>>>(in1, ws_packw, ws_wsq, ws_idx);
    }
}
